// NegLogLikelihoodFwdLoss_58609123721733
// MI455X (gfx1250) — matrix operand order repaired and checked by reading, not compiled, not run on hardware
//
#include <hip/hip_runtime.h>
#include <hip/hip_bf16.h>

typedef __attribute__((ext_vector_type(16))) _Float16 v16h;
typedef __attribute__((ext_vector_type(8)))  float    v8f;

#define CRF_B   256
#define CRF_T   2048
#define CRF_K2  52
#define START_TAG 50
#define STOP_TAG  51
#define ROWS    16
#define NEG_INF -10000.0f
#define LN2F     0.69314718055994531f
#define INV_LN2F 1.44269504088896340f

// One wave per block, 16 blocks; block g owns batch rows [16g,16g+16).
//
// Recursion kept in LOG2 domain, stored RELATIVE to an accumulated per-row
// shift Macc (log2 units): alphaRel[b,n] in LDS; per step
//   mx   = rowmax(alphaRel);  Macc += mx
//   eA   = exp2(alphaRel - mx)                      (raw v_exp_f32)
//   S    = eA @ exp(trans)^T                        (8x v_wmma f16->f32)
//   rel' = emit*(1/ln2) + log2(S)                   (raw v_log_f32 + fma)
// Absolute alpha2 = rel' + Macc is only needed at capture (t+1==len) and is
// reconstructed from register-resident fvRel + per-row Mcap at the end.
__global__ __launch_bounds__(32) void crf_fwd_kernel(
    const float* __restrict__ em,   // [B,T,K2]
    const float* __restrict__ tr,   // [K2,K2]  (tr[next*K2+prev])
    const int*   __restrict__ len,  // [B]
    const int*   __restrict__ lab,  // [B,T]
    float* __restrict__ partial)    // [16]
{
    __shared__ float alpha[ROWS][65];  // relative log2-domain alpha (padded)
    __shared__ float fvs[ROWS][65];

    const int lane   = threadIdx.x;        // 0..31
    const int wgbase = blockIdx.x * ROWS;
    const int r      = lane & 15;          // A row / N column
    const int hi     = lane >> 4;
    const int rbase  = hi * 8;             // C/D rows for this lane

    // init alpha0 (absolute == relative, Macc starts at 0)
    if (lane < 16) {
        for (int j = 0; j < 65; ++j)
            alpha[r][j] = (j == START_TAG) ? 0.0f : NEG_INF;
    }

    // constant B fragments: Bm[p][n] = exp(tr[n*K2+p]), zero-padded to 64x64
    v16h Bf[4][2];
#pragma unroll
    for (int nt = 0; nt < 4; ++nt) {
        int n = nt * 16 + r;
#pragma unroll
        for (int c = 0; c < 2; ++c) {
#pragma unroll
            for (int e = 0; e < 16; ++e) {
                int p = 32 * c + (hi ? 8 : 0) + e + ((e >= 8) ? 8 : 0);
                float v = (n < CRF_K2 && p < CRF_K2) ? __expf(tr[n * CRF_K2 + p]) : 0.0f;
                Bf[nt][c][e] = (_Float16)v;
            }
        }
    }

    int lenr[8];
#pragma unroll
    for (int rr = 0; rr < 8; ++rr) lenr[rr] = len[wgbase + rbase + rr];
    const int Lown = len[wgbase + r];      // length of this lane's own row

    float fvRel[4][8];                     // register-resident capture
#pragma unroll
    for (int nt = 0; nt < 4; ++nt)
#pragma unroll
        for (int rr = 0; rr < 8; ++rr) fvRel[nt][rr] = 0.0f;

    float Macc = 0.0f;                     // accumulated shift for own row
    float Mcap = 0.0f;                     // Macc at capture time for own row

    // ==================== sequential forward recursion ====================
    for (int t = 0; t < CRF_T; ++t) {
        __builtin_amdgcn_wave_barrier();

        // hint-prefetch next step's emission rows
        if (t + 1 < CRF_T) {
            const float* pf =
                &em[((size_t)(wgbase + r) * CRF_T + (t + 1)) * CRF_K2 + hi * 26];
            __builtin_prefetch(pf, 0, 1);
        }

        // current-step emissions into registers (clamped col -> no predication)
        float emr[4][8];
#pragma unroll
        for (int nt = 0; nt < 4; ++nt) {
            int n  = nt * 16 + r;
            int nc = (n < CRF_K2) ? n : (CRF_K2 - 1);
#pragma unroll
            for (int rr = 0; rr < 8; ++rr) {
                int row = rbase + rr;
                emr[nt][rr] = em[((size_t)(wgbase + row) * CRF_T + t) * CRF_K2 + nc];
            }
        }

        // per-row max over cols 0..51 (2 lanes/row, 26 each, tree-reduced)
        int cb = hi * 26;
        float t1[13];
#pragma unroll
        for (int j = 0; j < 13; ++j)
            t1[j] = fmaxf(alpha[r][cb + 2 * j], alpha[r][cb + 2 * j + 1]);
        float t2[7];
#pragma unroll
        for (int j = 0; j < 6; ++j) t2[j] = fmaxf(t1[2 * j], t1[2 * j + 1]);
        t2[6] = t1[12];
        float t3a = fmaxf(t2[0], t2[1]), t3b = fmaxf(t2[2], t2[3]);
        float t3c = fmaxf(t2[4], t2[5]);
        float mx  = fmaxf(fmaxf(t3a, t3b), fmaxf(t3c, t2[6]));
        mx = fmaxf(mx, __shfl_xor(mx, 16));

        Macc += mx;
        if (t + 1 == Lown) Mcap = Macc;    // own-row capture of shift

        // A fragments: exp2(rel - mx), raw v_exp_f32
        v16h af[2];
#pragma unroll
        for (int c = 0; c < 2; ++c) {
#pragma unroll
            for (int e = 0; e < 16; ++e) {
                int k = 32 * c + (hi ? 8 : 0) + e + ((e >= 8) ? 8 : 0);
                af[c][e] = (_Float16)__builtin_amdgcn_exp2f(alpha[r][k] - mx);
            }
        }

        __builtin_amdgcn_wave_barrier();

        // 8 WMMAs + branch-free consume
#pragma unroll
        for (int nt = 0; nt < 4; ++nt) {
            v8f acc = {};
            acc = __builtin_amdgcn_wmma_f32_16x16x32_f16(
                false, af[0], false, Bf[nt][0], (short)0, acc, false, false);
            acc = __builtin_amdgcn_wmma_f32_16x16x32_f16(
                false, af[1], false, Bf[nt][1], (short)0, acc, false, false);
            int n = nt * 16 + r;
#pragma unroll
            for (int rr = 0; rr < 8; ++rr) {
                // rel' = emit/ln2 + log2(S);  S==0 for padded n -> -Inf (safe)
                float anew = fmaf(emr[nt][rr], INV_LN2F,
                                  __builtin_amdgcn_logf(acc[rr]));
                alpha[rbase + rr][n] = anew;
                fvRel[nt][rr] = (t + 1 == lenr[rr]) ? anew : fvRel[nt][rr];
            }
        }
    }

    __builtin_amdgcn_wave_barrier();

    // dump register-resident fv to LDS (relative, log2 units)
#pragma unroll
    for (int nt = 0; nt < 4; ++nt) {
        int n = nt * 16 + r;
#pragma unroll
        for (int rr = 0; rr < 8; ++rr) fvs[rbase + rr][n] = fvRel[nt][rr];
    }
    __builtin_amdgcn_wave_barrier();

    // terminal: forward_score[b] = logsumexp_n(alpha_fin[b,n] + tr[stop,n])
    float fsc = 0.0f;
    if (lane < 16) {
        float m2 = -3.0e38f;
        for (int n = 0; n < CRF_K2; ++n) {
            float v = LN2F * (fvs[lane][n] + Mcap) + tr[STOP_TAG * CRF_K2 + n];
            m2 = fmaxf(m2, v);
        }
        float ss = 0.0f;
        for (int n = 0; n < CRF_K2; ++n) {
            float v = LN2F * (fvs[lane][n] + Mcap) + tr[STOP_TAG * CRF_K2 + n];
            ss += __expf(v - m2);
        }
        fsc = m2 + __logf(ss);
    }

    // gold score, time range split across the two half-waves
    int b = wgbase + r;
    int L = len[b];                        // 1..T
    int js = hi ? (L >> 1) : 0;
    int je = hi ? L : (L >> 1);
    float es = 0.0f, ts = 0.0f;
    int pv = (js == 0) ? START_TAG : lab[(size_t)b * CRF_T + js - 1];
    for (int j = js; j < je; ++j) {
        int lj = lab[(size_t)b * CRF_T + j];
        ts += tr[lj * CRF_K2 + pv];
        es += em[((size_t)b * CRF_T + j) * CRF_K2 + lj];
        pv = lj;
    }
    if (hi) ts += tr[STOP_TAG * CRF_K2 + lab[(size_t)b * CRF_T + L - 1]];
    es += __shfl_xor(es, 16);
    ts += __shfl_xor(ts, 16);

    float diff = (lane < 16) ? (fsc - es - ts) : 0.0f;
#pragma unroll
    for (int o = 16; o > 0; o >>= 1) diff += __shfl_xor(diff, o);
    if (lane == 0) partial[blockIdx.x] = diff;
}

// deterministic final reduction: fixed-order sum of the 16 block partials
__global__ void crf_reduce_kernel(const float* __restrict__ partial,
                                  float* __restrict__ out)
{
    float s = 0.0f;
    for (int i = 0; i < 16; ++i) s += partial[i];
    out[0] = s * (1.0f / (float)CRF_B);
}

extern "C" void kernel_launch(void* const* d_in, const int* in_sizes, int n_in,
                              void* d_out, int out_size, void* d_ws, size_t ws_size,
                              hipStream_t stream) {
    (void)in_sizes; (void)n_in; (void)out_size; (void)ws_size;
    const float* em  = (const float*)d_in[0];   // emissions   [B,T,K2] f32
    const float* tr  = (const float*)d_in[1];   // transitions [K2,K2]  f32
    const int*   len = (const int*)d_in[2];     // lengths     [B]
    const int*   lab = (const int*)d_in[3];     // labels      [B,T]
    float* out     = (float*)d_out;
    float* partial = (float*)d_ws;              // 16 floats of scratch

    crf_fwd_kernel<<<CRF_B / ROWS, 32, 0, stream>>>(em, tr, len, lab, partial);
    crf_reduce_kernel<<<1, 1, 0, stream>>>(partial, out);
}
